// Encoder_79061757985232
// MI455X (gfx1250) — compile-verified, matrix operand order checked by reading, not run on hardware
//
#include <hip/hip_runtime.h>
#include <hip/hip_bf16.h>
#include <math.h>

// ---------------------------------------------------------------------------
// Transformer encoder layer for MI455X (gfx1250, wave32, WMMA).
// N=4096, D=512, H=8, E=64. All matmuls via v_wmma_f32_16x16x32_bf16.
// Fragments = two contiguous 16-byte runs per lane -> 2x b128 loads.
// GEMM waves compute 16x64 tiles (A-fragment reused across 4 wmma).
// Attention is flash-style with block-cooperative async LDS staging of K/V
// (global_load_async_to_lds_b128 + s_wait_asynccnt).
// ---------------------------------------------------------------------------

typedef __bf16 bf16;
typedef __bf16 v16bf __attribute__((ext_vector_type(16)));
typedef __bf16 v8bf  __attribute__((ext_vector_type(8)));
typedef float  v8f   __attribute__((ext_vector_type(8)));
typedef int    v4i   __attribute__((ext_vector_type(4)));

#define NTOK   4096
#define DMODEL 512
#define EHEAD  64
#define NHEAD  8
#define LN_EPS 1e-5f

#define GAS __attribute__((address_space(1)))
#define LAS __attribute__((address_space(3)))

#if defined(__gfx1250__) && __has_builtin(__builtin_amdgcn_global_load_async_to_lds_b128)
#define HAVE_ASYNC_LDS 1
#else
#define HAVE_ASYNC_LDS 0
#endif

__device__ __forceinline__ v8f wmma_bf16(v16bf a, v16bf b, v8f c) {
  return __builtin_amdgcn_wmma_f32_16x16x32_bf16(false, a, false, b, (short)0, c,
                                                 false, false);
}

// A-fragment (16x32 bf16) from row-major source, tile origin applied.
// Per ISA 7.12.2: lane holds row m=lane&15, K runs [kb,kb+8) and [16+kb,16+kb+8)
// with kb = (lane<16 ? 0 : 8)  ->  two contiguous 16-byte loads.
__device__ __forceinline__ v16bf load_frag_a(const bf16* __restrict__ src, int ld,
                                             int lane) {
  const int m  = lane & 15;
  const int kb = (lane >> 4) << 3;
  const bf16* r = src + (size_t)m * ld + kb;
  v8bf lo = *(const v8bf*)(r);
  v8bf hi = *(const v8bf*)(r + 16);
  return __builtin_shufflevector(lo, hi, 0, 1, 2, 3, 4, 5, 6, 7,
                                 8, 9, 10, 11, 12, 13, 14, 15);
}
// B-fragment from transposed storage Bt[N][K]: identical gather to A.
__device__ __forceinline__ v16bf load_frag_bt(const bf16* __restrict__ src, int ld,
                                              int lane) {
  return load_frag_a(src, ld, lane);
}

// 16-byte global->LDS stage (async if available).
__device__ __forceinline__ void stage16(void* lds, const void* g) {
#if HAVE_ASYNC_LDS
  void* gnc = const_cast<void*>(g);
  __builtin_amdgcn_global_load_async_to_lds_b128((GAS v4i*)gnc, (LAS v4i*)lds, 0, 0);
#else
  *(uint4*)lds = *(const uint4*)g;
#endif
}
__device__ __forceinline__ void wait_stage() {
#if HAVE_ASYNC_LDS
#if __has_builtin(__builtin_amdgcn_s_wait_asynccnt)
  __builtin_amdgcn_s_wait_asynccnt(0);
#else
  asm volatile("s_wait_asynccnt 0x0" ::: "memory");
#endif
#endif
}

// --- converts ---------------------------------------------------------------
__global__ void cvt_kernel(const float* __restrict__ src, bf16* __restrict__ dst,
                           int n) {
  int i = blockIdx.x * blockDim.x + threadIdx.x;
  if (i < n) dst[i] = (bf16)src[i];
}
// [R,C] f32 -> [C,R] bf16
__global__ void cvt_t_kernel(const float* __restrict__ src, bf16* __restrict__ dst,
                             int R, int C) {
  int i = blockIdx.x * blockDim.x + threadIdx.x;
  if (i < R * C) {
    int r = i / C, c = i - r * C;
    dst[(size_t)c * R + r] = (bf16)src[i];
  }
}
// wq/wk/wv [H,D,E] f32 -> [H,E,D] bf16
__global__ void cvt_hte_kernel(const float* __restrict__ src, bf16* __restrict__ dst) {
  int i = blockIdx.x * blockDim.x + threadIdx.x;  // over H*D*E
  int h = i >> 15;              // D*E = 32768
  int rem = i & 32767;
  int d = rem >> 6, e = rem & 63;
  dst[((size_t)h * EHEAD + e) * DMODEL + d] = (bf16)src[i];
}

// --- QKV projection ---------------------------------------------------------
// grid(N/16, 6), block 128. Wave handles one (mat,head) combo = blockIdx.y*4+wv
// and produces the full 16x64 tile: A-fragment reused across 4 wmma.
// Q,K stored [2][H][N][E]; V stored transposed [H][E][N] for the PV gemm.
__global__ void __launch_bounds__(128)
qkv_kernel(const bf16* __restrict__ xb, const bf16* __restrict__ wqt,
           const bf16* __restrict__ wkt, const bf16* __restrict__ wvt,
           bf16* __restrict__ qk, bf16* __restrict__ vt) {
  const int tid = threadIdx.x, lane = tid & 31, wv = tid >> 5;
  const int mt    = blockIdx.x;
  const int combo = blockIdx.y * 4 + wv;      // 0..23
  const int mat   = combo >> 3, h = combo & 7;
  const bf16* w = (mat == 0 ? wqt : mat == 1 ? wkt : wvt) +
                  (size_t)h * EHEAD * DMODEL;

  v8f c0 = {}, c1 = {}, c2 = {}, c3 = {};
  for (int k = 0; k < DMODEL; k += 32) {
    v16bf a = load_frag_a(xb + (size_t)(mt * 16) * DMODEL + k, DMODEL, lane);
    c0 = wmma_bf16(a, load_frag_bt(w + (size_t)(0 * 16) * DMODEL + k, DMODEL, lane), c0);
    c1 = wmma_bf16(a, load_frag_bt(w + (size_t)(1 * 16) * DMODEL + k, DMODEL, lane), c1);
    c2 = wmma_bf16(a, load_frag_bt(w + (size_t)(2 * 16) * DMODEL + k, DMODEL, lane), c2);
    c3 = wmma_bf16(a, load_frag_bt(w + (size_t)(3 * 16) * DMODEL + k, DMODEL, lane), c3);
  }
  const int mb = (lane >> 4) << 3, nn = lane & 15;
  if (mat < 2) {
    bf16* dst = qk + (size_t)(mat * NHEAD + h) * NTOK * EHEAD +
                (size_t)(mt * 16) * EHEAD + nn;
#pragma unroll
    for (int v = 0; v < 8; ++v) {
      dst[(size_t)(mb + v) * EHEAD + 0]  = (bf16)c0[v];
      dst[(size_t)(mb + v) * EHEAD + 16] = (bf16)c1[v];
      dst[(size_t)(mb + v) * EHEAD + 32] = (bf16)c2[v];
      dst[(size_t)(mb + v) * EHEAD + 48] = (bf16)c3[v];
    }
  } else {
    bf16* dst = vt + (size_t)h * EHEAD * NTOK + (size_t)nn * NTOK + mt * 16 + mb;
#pragma unroll
    for (int v = 0; v < 8; ++v) {
      dst[(size_t)(0 * 16) * NTOK + v] = (bf16)c0[v];
      dst[(size_t)(1 * 16) * NTOK + v] = (bf16)c1[v];
      dst[(size_t)(2 * 16) * NTOK + v] = (bf16)c2[v];
      dst[(size_t)(3 * 16) * NTOK + v] = (bf16)c3[v];
    }
  }
}

// --- Flash attention --------------------------------------------------------
// grid(N/64, H), block 128. Block stages a 32-key K block (32x64) and Vt block
// (64x32) in LDS (async), each of the 4 waves owns a 16-row query tile.
__global__ void __launch_bounds__(128)
attn_kernel(const bf16* __restrict__ qk, const bf16* __restrict__ vt,
            bf16* __restrict__ zc) {
  const int tid = threadIdx.x, lane = tid & 31, wv = tid >> 5;
  const int h  = blockIdx.y;
  const int q0 = (blockIdx.x * 4 + wv) * 16;

  __shared__ alignas(16) bf16 kbuf[32 * EHEAD];      // [key][feat]
  __shared__ alignas(16) bf16 vbuf[EHEAD * 32];      // [e][key]
  __shared__ alignas(16) bf16 pbuf[4][16 * 32];      // per-wave P bounce

  const bf16* qptr = qk + (size_t)(0 * NHEAD + h) * NTOK * EHEAD;
  const bf16* kptr = qk + (size_t)(1 * NHEAD + h) * NTOK * EHEAD;
  const bf16* vth  = vt + (size_t)h * EHEAD * NTOK;

  const v16bf qa0 = load_frag_a(qptr + (size_t)q0 * EHEAD + 0, EHEAD, lane);
  const v16bf qa1 = load_frag_a(qptr + (size_t)q0 * EHEAD + 32, EHEAD, lane);

  v8f o0 = {}, o1 = {}, o2 = {}, o3 = {};
  float mrun[8], lrun[8];
#pragma unroll
  for (int v = 0; v < 8; ++v) { mrun[v] = -3.0e38f; lrun[v] = 0.f; }

  const int mb = (lane >> 4) << 3, nn = lane & 15;
  bf16* pw = &pbuf[wv][0];

  for (int j = 0; j < NTOK / 32; ++j) {
    // ---- stage K (4KB contiguous) and Vt (64 rows x 64B) into LDS ----
    const bf16* kg = kptr + (size_t)(j * 32) * EHEAD;
#pragma unroll
    for (int p = 0; p < 2; ++p) {
      int idx = p * 128 + tid;                       // 16B chunk index, 0..255
      stage16((uint4*)kbuf + idx, (const uint4*)kg + idx);
      int vr = idx >> 2, vc = idx & 3;
      stage16((uint4*)vbuf + idx,
              (const uint4*)(vth + (size_t)vr * NTOK + j * 32 + vc * 8));
    }
    if (j + 1 < NTOK / 32) {  // prefetch next block
      __builtin_prefetch(kptr + (size_t)(j + 1) * 32 * EHEAD, 0, 0);
      __builtin_prefetch(vth + (size_t)(j + 1) * 32, 0, 0);
    }
    wait_stage();
    __syncthreads();

    // ---- S = Q @ K^T (16x32 scores), scale, online softmax ----
    v8f s0 = {}, s1 = {};
    s0 = wmma_bf16(qa0, load_frag_bt(kbuf + 0, EHEAD, lane), s0);
    s0 = wmma_bf16(qa1, load_frag_bt(kbuf + 32, EHEAD, lane), s0);
    s1 = wmma_bf16(qa0, load_frag_bt(kbuf + 16 * EHEAD + 0, EHEAD, lane), s1);
    s1 = wmma_bf16(qa1, load_frag_bt(kbuf + 16 * EHEAD + 32, EHEAD, lane), s1);

    float p0[8], p1[8], sc[8];
#pragma unroll
    for (int v = 0; v < 8; ++v) {
      float a = s0[v] * 0.125f;  // 1/sqrt(E)
      float b = s1[v] * 0.125f;
      float rm = fmaxf(a, b);
#pragma unroll
      for (int mk = 1; mk < 16; mk <<= 1) rm = fmaxf(rm, __shfl_xor(rm, mk, 32));
      float mnew = fmaxf(mrun[v], rm);
      sc[v] = __expf(mrun[v] - mnew);
      p0[v] = __expf(a - mnew);
      p1[v] = __expf(b - mnew);
      float rs = p0[v] + p1[v];
#pragma unroll
      for (int mk = 1; mk < 16; mk <<= 1) rs += __shfl_xor(rs, mk, 32);
      lrun[v] = lrun[v] * sc[v] + rs;
      mrun[v] = mnew;
    }
#pragma unroll
    for (int v = 0; v < 8; ++v) {
      o0[v] *= sc[v]; o1[v] *= sc[v]; o2[v] *= sc[v]; o3[v] *= sc[v];
      pw[(mb + v) * 32 + nn]      = (bf16)p0[v];
      pw[(mb + v) * 32 + 16 + nn] = (bf16)p1[v];
    }
    // ---- O += P @ V  (V from transposed LDS tile) ----
    v16bf pa = load_frag_a(pw, 32, lane);   // wave-local RAW; DScnt-enforced
    o0 = wmma_bf16(pa, load_frag_bt(vbuf + 0 * 16 * 32, 32, lane), o0);
    o1 = wmma_bf16(pa, load_frag_bt(vbuf + 1 * 16 * 32, 32, lane), o1);
    o2 = wmma_bf16(pa, load_frag_bt(vbuf + 2 * 16 * 32, 32, lane), o2);
    o3 = wmma_bf16(pa, load_frag_bt(vbuf + 3 * 16 * 32, 32, lane), o3);
    __syncthreads();  // protect kbuf/vbuf before next stage
  }
#pragma unroll
  for (int v = 0; v < 8; ++v) {
    float inv = 1.0f / lrun[v];
    bf16* zr = zc + (size_t)(q0 + mb + v) * DMODEL + h * EHEAD;
    zr[0 * 16 + nn] = (bf16)(o0[v] * inv);
    zr[1 * 16 + nn] = (bf16)(o1[v] * inv);
    zr[2 * 16 + nn] = (bf16)(o2[v] * inv);
    zr[3 * 16 + nn] = (bf16)(o3[v] * inv);
  }
}

// --- GEMM: C(f32,[M,N]) = A(bf16,[M,K]) @ Bt(bf16,[N,K])^T -----------------
// grid(M/16, N/256), block 128. Each wave computes a 16x64 tile: the A
// fragment is loaded once per k-step and reused across 4 wmma.
__global__ void __launch_bounds__(128)
gemm_bt_kernel(const bf16* __restrict__ A, const bf16* __restrict__ Bt,
               float* __restrict__ C, int Kdim, int Ndim) {
  const int tid = threadIdx.x, lane = tid & 31, wv = tid >> 5;
  const int mt = blockIdx.x;
  const int n0 = (blockIdx.y * 4 + wv) * 64;
  v8f c0 = {}, c1 = {}, c2 = {}, c3 = {};
  for (int k = 0; k < Kdim; k += 32) {
    v16bf a = load_frag_a(A + (size_t)(mt * 16) * Kdim + k, Kdim, lane);
    c0 = wmma_bf16(a, load_frag_bt(Bt + (size_t)(n0 + 0)  * Kdim + k, Kdim, lane), c0);
    c1 = wmma_bf16(a, load_frag_bt(Bt + (size_t)(n0 + 16) * Kdim + k, Kdim, lane), c1);
    c2 = wmma_bf16(a, load_frag_bt(Bt + (size_t)(n0 + 32) * Kdim + k, Kdim, lane), c2);
    c3 = wmma_bf16(a, load_frag_bt(Bt + (size_t)(n0 + 48) * Kdim + k, Kdim, lane), c3);
  }
  const int mb = (lane >> 4) << 3, nn = lane & 15;
  float* dst = C + (size_t)(mt * 16) * Ndim + n0 + nn;
#pragma unroll
  for (int v = 0; v < 8; ++v) {
    dst[(size_t)(mb + v) * Ndim + 0]  = c0[v];
    dst[(size_t)(mb + v) * Ndim + 16] = c1[v];
    dst[(size_t)(mb + v) * Ndim + 32] = c2[v];
    dst[(size_t)(mb + v) * Ndim + 48] = c3[v];
  }
}

// --- Row LayerNorm: out = LN(t + resid (+ bias)) * g + b --------------------
__global__ void __launch_bounds__(128)
ln_kernel(const float* __restrict__ t, const float* __restrict__ resid,
          const float* __restrict__ bias, const float* __restrict__ g,
          const float* __restrict__ bvec, float* __restrict__ outf,
          bf16* __restrict__ outb) {
  const int tid = threadIdx.x, lane = tid & 31, wv = tid >> 5;
  const int row = blockIdx.x * (blockDim.x >> 5) + wv;
  const float* tr = t + (size_t)row * DMODEL;
  const float* rr = resid + (size_t)row * DMODEL;
  float y[16];
  float s = 0.f;
#pragma unroll
  for (int i = 0; i < 16; ++i) {
    int c = lane + 32 * i;
    float v = tr[c] + rr[c];
    if (bias) v += bias[c];
    y[i] = v;
    s += v;
  }
#pragma unroll
  for (int mk = 1; mk < 32; mk <<= 1) s += __shfl_xor(s, mk, 32);
  float mean = s * (1.0f / DMODEL);
  float q = 0.f;
#pragma unroll
  for (int i = 0; i < 16; ++i) { float d = y[i] - mean; q += d * d; }
#pragma unroll
  for (int mk = 1; mk < 32; mk <<= 1) q += __shfl_xor(q, mk, 32);
  float inv = rsqrtf(q * (1.0f / DMODEL) + LN_EPS);
#pragma unroll
  for (int i = 0; i < 16; ++i) {
    int c = lane + 32 * i;
    float o = (y[i] - mean) * inv * g[c] + bvec[c];
    outf[(size_t)row * DMODEL + c] = o;
    if (outb) outb[(size_t)row * DMODEL + c] = (bf16)o;
  }
}

// ---------------------------------------------------------------------------
extern "C" void kernel_launch(void* const* d_in, const int* in_sizes, int n_in,
                              void* d_out, int out_size, void* d_ws, size_t ws_size,
                              hipStream_t stream) {
  const float* x     = (const float*)d_in[0];
  const float* wq    = (const float*)d_in[1];
  const float* wk    = (const float*)d_in[2];
  const float* wv    = (const float*)d_in[3];
  const float* w_mh  = (const float*)d_in[4];
  const float* ln1_g = (const float*)d_in[5];
  const float* ln1_b = (const float*)d_in[6];
  const float* ffn_w = (const float*)d_in[7];
  const float* ffn_b = (const float*)d_in[8];
  const float* ln2_g = (const float*)d_in[9];
  const float* ln2_b = (const float*)d_in[10];
  float* out = (float*)d_out;

  char* ws = (char*)d_ws;
  size_t off = 0;
  auto carve = [&](size_t bytes) {
    void* p = ws + off;
    off = (off + bytes + 255) & ~(size_t)255;
    return p;
  };
  bf16*  xb   = (bf16*)carve((size_t)NTOK * DMODEL * 2);
  bf16*  wqt  = (bf16*)carve((size_t)NHEAD * DMODEL * EHEAD * 2);  // [H,E,D]
  bf16*  wkt  = (bf16*)carve((size_t)NHEAD * DMODEL * EHEAD * 2);
  bf16*  wvt  = (bf16*)carve((size_t)NHEAD * DMODEL * EHEAD * 2);
  bf16*  wmht = (bf16*)carve((size_t)DMODEL * DMODEL * 2);         // [n,k]
  bf16*  ffnb = (bf16*)carve((size_t)DMODEL * DMODEL * 2);         // already [n,k]
  bf16*  qkb  = (bf16*)carve((size_t)2 * NHEAD * NTOK * EHEAD * 2);
  bf16*  vtb  = (bf16*)carve((size_t)NHEAD * EHEAD * NTOK * 2);    // [H,E,N]
  bf16*  zc   = (bf16*)carve((size_t)NTOK * DMODEL * 2);
  float* t1   = (float*)carve((size_t)NTOK * DMODEL * 4);
  float* nzf  = (float*)carve((size_t)NTOK * DMODEL * 4);
  bf16*  nzb  = (bf16*)carve((size_t)NTOK * DMODEL * 2);
  float* t2   = (float*)carve((size_t)NTOK * DMODEL * 4);

  {
    int nx = NTOK * DMODEL, nw = NHEAD * DMODEL * EHEAD, nd = DMODEL * DMODEL;
    cvt_kernel<<<(nx + 255) / 256, 256, 0, stream>>>(x, xb, nx);
    cvt_hte_kernel<<<(nw + 255) / 256, 256, 0, stream>>>(wq, wqt);
    cvt_hte_kernel<<<(nw + 255) / 256, 256, 0, stream>>>(wk, wkt);
    cvt_hte_kernel<<<(nw + 255) / 256, 256, 0, stream>>>(wv, wvt);
    cvt_t_kernel<<<(nd + 255) / 256, 256, 0, stream>>>(w_mh, wmht, DMODEL, DMODEL);
    cvt_kernel<<<(nd + 255) / 256, 256, 0, stream>>>(ffn_w, ffnb, nd);
  }
  qkv_kernel<<<dim3(NTOK / 16, 6), 128, 0, stream>>>(xb, wqt, wkt, wvt, qkb, vtb);
  attn_kernel<<<dim3(NTOK / 64, NHEAD), 128, 0, stream>>>(qkb, vtb, zc);
  gemm_bt_kernel<<<dim3(NTOK / 16, DMODEL / 256), 128, 0, stream>>>(zc, wmht, t1,
                                                                    DMODEL, DMODEL);
  ln_kernel<<<NTOK / 4, 128, 0, stream>>>(t1, x, nullptr, ln1_g, ln1_b, nzf, nzb);
  gemm_bt_kernel<<<dim3(NTOK / 16, DMODEL / 256), 128, 0, stream>>>(nzb, ffnb, t2,
                                                                    DMODEL, DMODEL);
  ln_kernel<<<NTOK / 4, 128, 0, stream>>>(t2, nzf, ffn_b, ln2_g, ln2_b, out, nullptr);
}
